// MarginLoss_34565896798668
// MI455X (gfx1250) — matrix operand order repaired and checked by reading, not compiled, not run on hardware
//
#include <hip/hip_runtime.h>
#include <hip/hip_bf16.h>
#include <hip/hip_fp16.h>
#include <stdint.h>

typedef __attribute__((ext_vector_type(16))) _Float16 v16h;
typedef __attribute__((ext_vector_type(8)))  _Float16 v8h;
typedef __attribute__((ext_vector_type(8)))  float    v8f;

#define NROW   8192
#define DIM    512
#define MARGIN 0.2f
#define EPSV   1e-8f

#define BT     128              // output tile (BT x BT)
#define KC     32               // k-chunk staged in LDS per stage (1 WMMA step)
#define LDT    40               // LDS row stride in f16 (32 + 8 pad, 80B = 5*16B)
#define NSTAGE (DIM / KC)       // 16
#define TILES  (NROW / BT)      // 64

// ---------------------------------------------------------------------------
// CDNA5 async global->LDS copy (16B per lane), tracked by ASYNCcnt.
// ---------------------------------------------------------------------------
__device__ __forceinline__ void async_copy_b128(uint32_t lds_byte_addr,
                                                uint64_t gaddr) {
  asm volatile("global_load_async_to_lds_b128 %0, %1, off"
               :: "v"(lds_byte_addr), "v"(gaddr)
               : "memory");
}
__device__ __forceinline__ void wait_async_le4() {
  asm volatile("s_wait_asynccnt 0x4" ::: "memory");
}
__device__ __forceinline__ void wait_async_zero() {
  asm volatile("s_wait_asynccnt 0x0" ::: "memory");
}

// ---------------------------------------------------------------------------
// Kernel 1: row-normalize f32 -> f16  (one block of 256 threads per row)
// ---------------------------------------------------------------------------
__global__ __launch_bounds__(256) void normalize_rows_f16(
    const float* __restrict__ x, _Float16* __restrict__ en) {
  const int row = blockIdx.x;
  const int t = threadIdx.x;
  const float* xr = x + (size_t)row * DIM;
  const float a = xr[t];
  const float b = xr[t + 256];
  float ss = a * a + b * b;
  #pragma unroll
  for (int off = 16; off; off >>= 1) ss += __shfl_xor(ss, off, 32);
  __shared__ float wsum[8];
  if ((t & 31) == 0) wsum[t >> 5] = ss;
  __syncthreads();
  float tot = 0.f;
  #pragma unroll
  for (int i = 0; i < 8; ++i) tot += wsum[i];
  const float scale = 1.0f / fmaxf(sqrtf(tot), EPSV);
  _Float16* er = en + (size_t)row * DIM;
  er[t]       = (_Float16)(a * scale);
  er[t + 256] = (_Float16)(b * scale);
}

// ---------------------------------------------------------------------------
// Kernel 2: 128x128 tile of sim = En * En^T via v_wmma_f32_16x16x32_f16.
// Async global->LDS double-buffered staging, fused margin-loss epilogue,
// lower-triangle tiles only (off-diagonal weighted x2).
// ---------------------------------------------------------------------------
__global__ __launch_bounds__(256) void sim_loss_tile(
    const _Float16* __restrict__ en, const int* __restrict__ labels,
    float* __restrict__ partials) {
  const int bx = blockIdx.x;   // tile row
  const int by = blockIdx.y;   // tile col
  const int t  = threadIdx.x;
  const int pidx = bx * TILES + by;
  if (by > bx) {               // upper triangle: covered by symmetry
    if (t == 0) partials[pidx] = 0.0f;
    return;
  }

  __shared__ _Float16 As[2][BT * LDT];
  __shared__ _Float16 Bs[2][BT * LDT];
  __shared__ int   labR[BT];
  __shared__ int   labC[BT];
  __shared__ float redbuf[8];

  const int rowBase = bx * BT;
  const int colBase = by * BT;

  // Per-thread staging coordinates: 2 b128 chunks per tile per stage.
  // idx = i*256 + t -> row = idx>>2 (0..127), chunk col = (idx&3)*8 f16.
  int srow[2], scol[2];
  #pragma unroll
  for (int i = 0; i < 2; ++i) {
    const int idx = i * 256 + t;
    srow[i] = idx >> 2;
    scol[i] = (idx & 3) * 8;
  }

  // Issue one stage of async copies into buffer b.
  auto issue_stage = [&](int kc, int b) {
    #pragma unroll
    for (int i = 0; i < 2; ++i) {
      const int r = srow[i];
      const int c = scol[i];
      async_copy_b128(
          (uint32_t)(uintptr_t)&As[b][r * LDT + c],
          (uint64_t)(uintptr_t)&en[(size_t)(rowBase + r) * DIM + kc + c]);
      async_copy_b128(
          (uint32_t)(uintptr_t)&Bs[b][r * LDT + c],
          (uint64_t)(uintptr_t)&en[(size_t)(colBase + r) * DIM + kc + c]);
    }
  };

  if (t < BT) labR[t]      = labels[rowBase + t];
  else        labC[t - BT] = labels[colBase + (t - BT)];

  const int lane = t & 31;
  const int wave = t >> 5;          // 0..7
  const int wr   = wave >> 2;       // 0..1 : wave row  (64 rows each)
  const int wc   = wave & 3;        // 0..3 : wave col  (32 cols each)
  const int half = lane >> 4;       // lane group 0/1
  const int mrow = lane & 15;

  v8f acc[4][2] = {};               // 4x2 fragments of 16x16 f32

  issue_stage(0, 0);                // prologue

  for (int s = 0; s < NSTAGE; ++s) {
    const int cur = s & 1;
    if (s + 1 < NSTAGE) {
      issue_stage((s + 1) * KC, cur ^ 1);
      wait_async_le4();             // oldest 4 (stage s) have landed in LDS
    } else {
      wait_async_zero();
    }
    __syncthreads();                // stage s visible to all waves

    const _Float16* Ab = As[cur];
    const _Float16* Bb = Bs[cur];

    // B fragments: lane holds col n = mrow, K = 16*half .. +15 (32B contig)
    v16h bf[2];
    #pragma unroll
    for (int j = 0; j < 2; ++j) {
      const int n = wc * 32 + j * 16 + mrow;
      const _Float16* p = &Bb[n * LDT + half * 16];
      v8h lo = *(const v8h*)p;
      v8h hi = *(const v8h*)(p + 8);
      bf[j] = __builtin_shufflevector(lo, hi, 0,1,2,3,4,5,6,7,
                                      8,9,10,11,12,13,14,15);
    }
    // A fragments: lane holds row m, K = 16*half .. +15 (32B contig)
    #pragma unroll
    for (int i = 0; i < 4; ++i) {
      const int m = wr * 64 + i * 16 + mrow;
      const _Float16* p = &Ab[m * LDT + half * 16];
      v8h lo = *(const v8h*)p;
      v8h hi = *(const v8h*)(p + 8);
      v16h af = __builtin_shufflevector(lo, hi, 0,1,2,3,4,5,6,7,
                                        8,9,10,11,12,13,14,15);
      #pragma unroll
      for (int j = 0; j < 2; ++j) {
        acc[i][j] = __builtin_amdgcn_wmma_f32_16x16x32_f16(
            false, af, false, bf[j], (short)0, acc[i][j], false, false);
      }
    }
    __syncthreads();                // all waves done reading buf[cur]
  }

  // Fused loss epilogue. C/D layout: VGPR r -> M = r + 8*half, N = lane&15.
  float lsum = 0.0f;
  #pragma unroll
  for (int i = 0; i < 4; ++i) {
    const int fragRow = wr * 64 + i * 16;
    #pragma unroll
    for (int j = 0; j < 2; ++j) {
      const int fragCol = wc * 32 + j * 16;
      const int lc = labC[fragCol + mrow];
      #pragma unroll
      for (int r = 0; r < 8; ++r) {
        const int ridx = fragRow + r + 8 * half;
        const float s = acc[i][j][r];
        const bool same = (labR[ridx] == lc);
        lsum += same ? (1.0f - s) : fmaxf(s - MARGIN, 0.0f);
      }
    }
  }

  // Deterministic block reduction; off-diagonal tiles count twice (symmetry).
  #pragma unroll
  for (int off = 16; off; off >>= 1) lsum += __shfl_xor(lsum, off, 32);
  if (lane == 0) redbuf[wave] = lsum;
  __syncthreads();
  if (t == 0) {
    float tot = 0.0f;
    #pragma unroll
    for (int i = 0; i < 8; ++i) tot += redbuf[i];
    const float wgt = (bx == by) ? 1.0f : 2.0f;
    partials[pidx] = wgt * tot;
  }
}

// ---------------------------------------------------------------------------
// Kernel 3: deterministic fixed-order reduction of the 4096 tile partials.
// ---------------------------------------------------------------------------
__global__ __launch_bounds__(256) void final_reduce(
    const float* __restrict__ partials, float* __restrict__ out) {
  const int t = threadIdx.x;
  double s = 0.0;
  for (int i = t; i < TILES * TILES; i += 256) s += (double)partials[i];
  #pragma unroll
  for (int off = 16; off; off >>= 1) s += __shfl_xor(s, off, 32);
  __shared__ double buf[8];
  if ((t & 31) == 0) buf[t >> 5] = s;
  __syncthreads();
  if (t == 0) {
    double tot = 0.0;
    #pragma unroll
    for (int i = 0; i < 8; ++i) tot += buf[i];
    out[0] = (float)(tot / ((double)NROW * (double)NROW));
  }
}

// ---------------------------------------------------------------------------
extern "C" void kernel_launch(void* const* d_in, const int* in_sizes, int n_in,
                              void* d_out, int out_size, void* d_ws,
                              size_t ws_size, hipStream_t stream) {
  (void)in_sizes; (void)n_in; (void)out_size; (void)ws_size;
  const float* embeds = (const float*)d_in[0];
  const int*   labels = (const int*)d_in[1];
  float*       out    = (float*)d_out;

  _Float16* en = (_Float16*)d_ws;                       // 8192*512*2 = 8 MB
  float* partials =
      (float*)((char*)d_ws + (size_t)NROW * DIM * sizeof(_Float16));

  normalize_rows_f16<<<NROW, 256, 0, stream>>>(embeds, en);
  dim3 grid(TILES, TILES);
  sim_loss_tile<<<grid, 256, 0, stream>>>(en, labels, partials);
  final_reduce<<<1, 256, 0, stream>>>(partials, out);
}
